// Network_36472862278367
// MI455X (gfx1250) — compile-verified, matrix operand order checked by reading, not run on hardware
//
#include <hip/hip_runtime.h>
#include <hip/hip_bf16.h>

// ---------------- CDNA5 WMMA plumbing (wave32, 16x16x32 bf16) ----------------
typedef __attribute__((ext_vector_type(16))) __bf16 v16bf;
typedef __attribute__((ext_vector_type(8)))  float  v8f;

// Native bf16 convert (gfx1250 has bf16 VALU/cvt; avoids 4-op software RNE chain)
__device__ __forceinline__ __bf16 tobf(float f) { return (__bf16)f; }

__device__ __forceinline__ v8f wmma_bf16(v16bf a, v16bf b, v8f c) {
  return __builtin_amdgcn_wmma_f32_16x16x32_bf16(
      /*neg_a=*/false, a, /*neg_b=*/false, b,
      /*c_mod=*/(short)0, c, /*reuse_a=*/false, /*reuse_b=*/false);
}

// A fragment: 16x32 bf16, row-major source, stride lda (elements).
// ISA layout: lanes 0-15 -> M=lane, K in {0..7,16..23}; lanes 16-31 -> M=lane-16, K in {8..15,24..31}
// Per lane this is two contiguous 8-element runs -> compiler merges to 2x b128.
__device__ __forceinline__ v16bf load_a(const __bf16* A, long lda, int valid_rows) {
  int lane = threadIdx.x & 31;
  int half = lane >> 4, m = lane & 15;
  v16bf a;
  if (m < valid_rows) {
    const __bf16* row = A + (long)m * lda + (half << 3);
#pragma unroll
    for (int i = 0; i < 8; ++i) a[i]     = row[i];        // K = half*8 + 0..7
#pragma unroll
    for (int i = 0; i < 8; ++i) a[8 + i] = row[16 + i];   // K = 16 + half*8 + 0..7
  } else {
#pragma unroll
    for (int i = 0; i < 16; ++i) a[i] = (__bf16)0.0f;
  }
  return a;
}

// Packed-B: each 32(K)x16(N) tile stored as 32 lanes x 16 bf16 (32B contiguous per lane).
__device__ __forceinline__ v16bf load_bp(const __bf16* tile) {
  int lane = threadIdx.x & 31;
  return *(const v16bf*)(tile + (lane << 4));
}

// Map logical B element (k, j) of a (K x N) matrix to packed fragment-tile index.
__device__ __forceinline__ long pack_idx(int k, int j, int N) {
  int kt = k >> 5, kl = k & 31;
  int nt = j >> 4, n = j & 15;
  int half = (kl >> 3) & 1;
  int base = kl & ~8;               // {0..7, 16..23}
  int b = base & 1;
  int q = base >> 1;                // {0..3, 8..11}
  int p = (q & 3) | ((q & 8) >> 1); // 0..7
  int e = 2 * p + b;                // element within lane
  int lane = half * 16 + n;
  return (((long)(kt * (N >> 4) + nt) * 32 + lane) << 4) + e;
}

// ---------------- problem constants ----------------
#define BB 16
#define SS 512
#define NA 6
#define GLOB 30
#define MASK 52
#define AFEAT 73
#define OBS 102
#define OBSP 128
#define EMB 512
#define HH 1024
#define HEADS 4
#define G0O 256
#define G1O 128
#define ACTH 256
#define ACTD 52
#define ACTDP 64
#define INW (GLOB + NA * AFEAT + MASK)   /* 520 */
#define ROWS (BB * NA)                   /* 96  */
#define TILE 512                          /* bf16 elements per packed 32x16 tile */

// ================= prep / packing kernels (one-time, scatter is fine) =================
__global__ void k_prep_vec(const float* __restrict__ in, __bf16* __restrict__ vec) {
  long i = (long)blockIdx.x * blockDim.x + threadIdx.x;
  long total = (long)ROWS * SS * OBSP;
  if (i >= total) return;
  int c = (int)(i & (OBSP - 1));
  long rs = i >> 7;
  int s = (int)(rs % SS);
  int r = (int)(rs / SS);
  int b = r / NA, a = r % NA;
  float v = 0.f;
  const float* row = in + ((long)b * SS + s) * INW;
  if (c < 72)       v = row[GLOB + a * AFEAT + (c + 1)];   // drop agent feature 0
  else if (c < OBS) v = row[c - 72];                       // global features
  vec[i] = tobf(v);
}

// emb_W (5,102,512) -> per-type packed B (K=128 padded, N=512)
__global__ void k_pack_embW(const float* __restrict__ w, __bf16* __restrict__ out) {
  long i = (long)blockIdx.x * blockDim.x + threadIdx.x;
  long total = 5L * OBSP * EMB;
  if (i >= total) return;
  int o = (int)(i % EMB);
  int k = (int)((i / EMB) % OBSP);
  int t = (int)(i / ((long)EMB * OBSP));
  __bf16 v = (k < OBS) ? tobf(w[((long)t * OBS + k) * EMB + o]) : (__bf16)0.0f;
  out[(long)t * OBSP * EMB + pack_idx(k, o, EMB)] = v;
}

// W (G,K) row-major -> packed B = W^T (K x G)
__global__ void k_pack_T(const float* __restrict__ in, __bf16* __restrict__ out, int G, int K) {
  long i = (long)blockIdx.x * blockDim.x + threadIdx.x;
  if (i >= (long)G * K) return;
  int g = (int)(i / K), k = (int)(i % K);
  out[pack_idx(k, g, G)] = tobf(in[i]);
}

// (Hn,K,O) -> packed B (K x Hn*O)
__global__ void k_pack_heads(const float* __restrict__ in, __bf16* __restrict__ out, int Hn, int K, int O) {
  long i = (long)blockIdx.x * blockDim.x + threadIdx.x;
  if (i >= (long)Hn * K * O) return;
  int o = (int)(i % O);
  long hk = i / O;
  int k = (int)(hk % K);
  int h = (int)(hk / K);
  out[pack_idx(k, h * O + o, Hn * O)] = tobf(in[i]);
}

// aW1 (5,512,256) -> per-type packed B (512 x 256)
__global__ void k_pack_aW1(const float* __restrict__ in, __bf16* __restrict__ out) {
  long i = (long)blockIdx.x * blockDim.x + threadIdx.x;
  long total = 5L * 512 * ACTH;
  if (i >= total) return;
  int o = (int)(i % ACTH);
  int k = (int)((i / ACTH) % 512);
  int t = (int)(i / ((long)ACTH * 512));
  out[(long)t * 512 * ACTH + pack_idx(k, o, ACTH)] = tobf(in[i]);
}

// aW2 (5,256,52) -> per-type packed B (256 x 64 padded)
__global__ void k_pack_aW2(const float* __restrict__ in, __bf16* __restrict__ out) {
  long i = (long)blockIdx.x * blockDim.x + threadIdx.x;
  long total = 5L * ACTH * ACTDP;
  if (i >= total) return;
  int o = (int)(i % ACTDP);
  int k = (int)((i / ACTDP) % ACTH);
  int t = (int)(i / ((long)ACTDP * ACTH));
  __bf16 v = (o < ACTD) ? tobf(in[((long)t * ACTH + k) * ACTD + o]) : (__bf16)0.0f;
  out[(long)t * ACTH * ACTDP + pack_idx(k, o, ACTDP)] = v;
}

__global__ void k_init_h(const float* __restrict__ state, float* __restrict__ hf, __bf16* __restrict__ hb) {
  long i = (long)blockIdx.x * blockDim.x + threadIdx.x;
  if (i >= (long)ROWS * HH) return;
  hf[i] = state[i];
  hb[i] = tobf(state[i]);
}

__global__ void k_copy_hfinal(const float* __restrict__ hf, float* __restrict__ out) {
  long i = (long)blockIdx.x * blockDim.x + threadIdx.x;
  if (i >= (long)ROWS * HH) return;
  out[i] = hf[i];
}

// ================= embedding: emb = relu(vec @ embW[type] + emb_b[type]) =================
__global__ void k_embed(const __bf16* __restrict__ vec, const __bf16* __restrict__ embWp,
                        const float* __restrict__ emb_b, const int* __restrict__ types,
                        __bf16* __restrict__ emb) {
  int r = blockIdx.x;
  int mt = blockIdx.y;
  int w = threadIdx.x >> 5, lane = threadIdx.x & 31;
  int nt = blockIdx.z * 4 + w;                 // 0..31
  int t = types[r];
  const __bf16* A = vec + ((long)r * SS + mt * 16) * OBSP;
  const __bf16* B = embWp + (long)t * OBSP * EMB;
  v8f acc = {};
#pragma unroll
  for (int kk = 0; kk < OBSP / 32; ++kk) {
    v16bf a = load_a(A + kk * 32, OBSP, 16);
    v16bf b = load_bp(B + ((long)kk * (EMB >> 4) + nt) * TILE);
    acc = wmma_bf16(a, b, acc);
  }
  int half = lane >> 4, n = lane & 15;
  int col = nt * 16 + n;
#pragma unroll
  for (int vr = 0; vr < 8; ++vr) {
    int m = vr + half * 8;
    int s = mt * 16 + m;
    float v = acc[vr] + emb_b[(long)t * EMB + col];
    v = v > 0.f ? v : 0.f;
    emb[((long)r * SS + s) * EMB + col] = tobf(v);
  }
}

// ================= fused GRU step (x-proj + h-proj + gating) =================
// grid (6, 16), block 128. wave computes one 16x16 tile of r,z,xn,hn over H=1024.
__global__ void k_gru_step(const __bf16* __restrict__ emb, const __bf16* __restrict__ WihP,
                           const __bf16* __restrict__ WhhP,
                           const float* __restrict__ b_ih, const float* __restrict__ b_hh,
                           const float* __restrict__ hpf, const __bf16* __restrict__ hpb,
                           float* __restrict__ hnf, __bf16* __restrict__ hnb,
                           __bf16* __restrict__ gnn, int t) {
  int w = threadIdx.x >> 5, lane = threadIdx.x & 31;
  int mt = blockIdx.x;                  // 0..5
  int nt = blockIdx.y * 4 + w;          // 0..63
  const int NT3 = (3 * HH) >> 4;        // 192 tiles across 3H
  v8f ar = {}, az = {}, axn = {}, ahn = {};

  // x part: A rows = emb[(mt*16+m), t, :]; K = 512
  const __bf16* Abase = emb + (long)(mt * 16) * SS * EMB + (long)t * EMB;
  for (int kk = 0; kk < EMB / 32; ++kk) {
    v16bf a  = load_a(Abase + kk * 32, (long)SS * EMB, 16);
    v16bf br = load_bp(WihP + ((long)kk * NT3 + nt) * TILE);
    v16bf bz = load_bp(WihP + ((long)kk * NT3 + 64 + nt) * TILE);
    v16bf bn = load_bp(WihP + ((long)kk * NT3 + 128 + nt) * TILE);
    ar  = wmma_bf16(a, br, ar);
    az  = wmma_bf16(a, bz, az);
    axn = wmma_bf16(a, bn, axn);
  }
  // h part: K = 1024
  const __bf16* Hbase = hpb + (long)(mt * 16) * HH;
  for (int kk = 0; kk < HH / 32; ++kk) {
    v16bf a  = load_a(Hbase + kk * 32, HH, 16);
    v16bf br = load_bp(WhhP + ((long)kk * NT3 + nt) * TILE);
    v16bf bz = load_bp(WhhP + ((long)kk * NT3 + 64 + nt) * TILE);
    v16bf bn = load_bp(WhhP + ((long)kk * NT3 + 128 + nt) * TILE);
    ar  = wmma_bf16(a, br, ar);
    az  = wmma_bf16(a, bz, az);
    ahn = wmma_bf16(a, bn, ahn);
  }

  int half = lane >> 4, n = lane & 15;
  int col = nt * 16 + n;
#pragma unroll
  for (int vr = 0; vr < 8; ++vr) {
    int m = vr + half * 8;
    int row = mt * 16 + m;
    float r = 1.f / (1.f + __expf(-(ar[vr] + b_ih[col] + b_hh[col])));
    float z = 1.f / (1.f + __expf(-(az[vr] + b_ih[HH + col] + b_hh[HH + col])));
    float nn = tanhf(axn[vr] + b_ih[2 * HH + col] + r * (ahn[vr] + b_hh[2 * HH + col]));
    float hp = hpf[(long)row * HH + col];
    float hv = (1.f - z) * nn + z * hp;
    hnf[(long)row * HH + col] = hv;
    hnb[(long)row * HH + col] = tobf(hv);
    int b = row / NA, ag = row % NA;
    gnn[(((long)b * SS + t) * NA + ag) * HH + col] = tobf(hv);  // (B,S,agent,H)
  }
}

// ================= GAT0 (fused, in-place on gnn buffer) =================
__global__ void k_gat0(__bf16* __restrict__ gnn, const __bf16* __restrict__ W0P,
                       const float* __restrict__ a0) {
  __shared__ __bf16 Wh[NA * HEADS * G0O];      // 6 x 1024
  __shared__ float eS[24], jS[24], attnS[144];
  long bs = blockIdx.x;
  __bf16* X = gnn + bs * (long)NA * HH;
  int w = threadIdx.x >> 5, lane = threadIdx.x & 31;
  int half = lane >> 4, ncol = lane & 15;
  const int NT = (HEADS * G0O) >> 4;           // 64

  for (int nt = w; nt < NT; nt += 8) {
    v8f acc = {};
    for (int kk = 0; kk < HH / 32; ++kk) {
      v16bf a = load_a(X + kk * 32, HH, NA);
      v16bf b = load_bp(W0P + ((long)kk * NT + nt) * TILE);
      acc = wmma_bf16(a, b, acc);
    }
#pragma unroll
    for (int vr = 0; vr < 8; ++vr) {
      int m = vr + half * 8;
      if (m < NA) Wh[m * (HEADS * G0O) + nt * 16 + ncol] = tobf(acc[vr]);
    }
  }
  __syncthreads();

  int tid = threadIdx.x;
  if (tid < 24) {                         // ei/ej per (head, agent)
    int k = tid / NA, nA = tid % NA;
    float ei = 0.f, ej = 0.f;
    for (int o = 0; o < G0O; ++o) {
      float wv = (float)Wh[nA * (HEADS * G0O) + k * G0O + o];
      ei += wv * a0[(long)k * (2 * G0O) + o];
      ej += wv * a0[(long)k * (2 * G0O) + G0O + o];
    }
    eS[tid] = ei; jS[tid] = ej;
  }
  __syncthreads();
  if (tid < 24) {                         // relu + softmax over m
    int k = tid / NA, nA = tid % NA;
    float e[NA], mx = -1e30f;
    for (int m = 0; m < NA; ++m) { float v = eS[k * NA + nA] + jS[k * NA + m]; v = v > 0.f ? v : 0.f; e[m] = v; mx = v > mx ? v : mx; }
    float s = 0.f;
    for (int m = 0; m < NA; ++m) { e[m] = __expf(e[m] - mx); s += e[m]; }
    for (int m = 0; m < NA; ++m) attnS[tid * NA + m] = e[m] / s;
  }
  __syncthreads();
  for (int idx = tid; idx < HEADS * NA * G0O; idx += blockDim.x) {
    int k = idx / (NA * G0O), rem = idx % (NA * G0O), nA = rem / G0O, o = rem % G0O;
    float v = 0.f;
    for (int m = 0; m < NA; ++m)
      v += attnS[(k * NA + nA) * NA + m] * (float)Wh[m * (HEADS * G0O) + k * G0O + o];
    v = v > 0.f ? v : (__expf(v) - 1.f);                  // ELU
    X[(long)nA * HH + k * G0O + o] = tobf(v);             // in-place: g0[b,s,n, k*256+o]
  }
}

// ================= GAT1 (fused, only agent-0 output -> feat0) =================
__global__ void k_gat1(const __bf16* __restrict__ g0, const __bf16* __restrict__ W1P,
                       const float* __restrict__ a1, __bf16* __restrict__ feat0) {
  __shared__ __bf16 Wh[NA * HEADS * G1O];      // 6 x 512
  __shared__ float eiS[4], ejS[24], attnS[24];
  long bs = blockIdx.x;
  const __bf16* X = g0 + bs * (long)NA * HH;
  int w = threadIdx.x >> 5, lane = threadIdx.x & 31;
  int half = lane >> 4, ncol = lane & 15;
  const int NT = (HEADS * G1O) >> 4;           // 32

  for (int nt = w; nt < NT; nt += 8) {
    v8f acc = {};
    for (int kk = 0; kk < HH / 32; ++kk) {
      v16bf a = load_a(X + kk * 32, HH, NA);
      v16bf b = load_bp(W1P + ((long)kk * NT + nt) * TILE);
      acc = wmma_bf16(a, b, acc);
    }
#pragma unroll
    for (int vr = 0; vr < 8; ++vr) {
      int m = vr + half * 8;
      if (m < NA) Wh[m * (HEADS * G1O) + nt * 16 + ncol] = tobf(acc[vr]);
    }
  }
  __syncthreads();
  int tid = threadIdx.x;
  if (tid < 24) {
    int k = tid / NA, mA = tid % NA;
    float ej = 0.f;
    for (int o = 0; o < G1O; ++o)
      ej += (float)Wh[mA * (HEADS * G1O) + k * G1O + o] * a1[(long)k * (2 * G1O) + G1O + o];
    ejS[tid] = ej;
  }
  if (tid < 4) {                          // ei for agent 0 only
    float ei = 0.f;
    for (int o = 0; o < G1O; ++o)
      ei += (float)Wh[0 * (HEADS * G1O) + tid * G1O + o] * a1[(long)tid * (2 * G1O) + o];
    eiS[tid] = ei;
  }
  __syncthreads();
  if (tid < 4) {
    float e[NA], mx = -1e30f;
    for (int m = 0; m < NA; ++m) { float v = eiS[tid] + ejS[tid * NA + m]; v = v > 0.f ? v : 0.f; e[m] = v; mx = v > mx ? v : mx; }
    float s = 0.f;
    for (int m = 0; m < NA; ++m) { e[m] = __expf(e[m] - mx); s += e[m]; }
    for (int m = 0; m < NA; ++m) attnS[tid * NA + m] = e[m] / s;
  }
  __syncthreads();
  for (int idx = tid; idx < HEADS * G1O; idx += blockDim.x) {
    int k = idx / G1O, o = idx % G1O;
    float v = 0.f;
    for (int m = 0; m < NA; ++m)
      v += attnS[k * NA + m] * (float)Wh[m * (HEADS * G1O) + k * G1O + o];
    v = v > 0.f ? v : (__expf(v) - 1.f);
    feat0[bs * (long)(HEADS * G1O) + idx] = tobf(v);
  }
}

// ================= actor head =================
__global__ void k_actor1(const __bf16* __restrict__ feat0, const __bf16* __restrict__ aW1p,
                         const float* __restrict__ ab1, const int* __restrict__ types,
                         __bf16* __restrict__ h1) {
  int b = blockIdx.x, mt = blockIdx.y;
  int w = threadIdx.x >> 5, lane = threadIdx.x & 31;
  int nt = blockIdx.z * 4 + w;          // 0..15
  int t = types[b * NA];
  const __bf16* A = feat0 + ((long)b * SS + mt * 16) * 512;
  const __bf16* B = aW1p + (long)t * 512 * ACTH;
  const int NT = ACTH >> 4;             // 16
  v8f acc = {};
  for (int kk = 0; kk < 512 / 32; ++kk) {
    v16bf a = load_a(A + kk * 32, 512, 16);
    v16bf bb = load_bp(B + ((long)kk * NT + nt) * TILE);
    acc = wmma_bf16(a, bb, acc);
  }
  int half = lane >> 4, n = lane & 15, col = nt * 16 + n;
#pragma unroll
  for (int vr = 0; vr < 8; ++vr) {
    int m = vr + half * 8, s = mt * 16 + m;
    float v = acc[vr] + ab1[(long)t * ACTH + col];
    v = v > 0.f ? v : 0.f;
    h1[((long)b * SS + s) * ACTH + col] = tobf(v);
  }
}

__global__ void k_actor2(const __bf16* __restrict__ h1, const __bf16* __restrict__ aW2p,
                         const float* __restrict__ ab2, const int* __restrict__ types,
                         const float* __restrict__ inputs, float* __restrict__ out) {
  int b = blockIdx.x, mt = blockIdx.y;
  int w = threadIdx.x >> 5, lane = threadIdx.x & 31;
  int nt = w;                            // 0..3 (N padded to 64)
  int t = types[b * NA];
  const __bf16* A = h1 + ((long)b * SS + mt * 16) * ACTH;
  const __bf16* B = aW2p + (long)t * ACTH * ACTDP;
  const int NT = ACTDP >> 4;             // 4
  v8f acc = {};
  for (int kk = 0; kk < ACTH / 32; ++kk) {
    v16bf a = load_a(A + kk * 32, ACTH, 16);
    v16bf bb = load_bp(B + ((long)kk * NT + nt) * TILE);
    acc = wmma_bf16(a, bb, acc);
  }
  int half = lane >> 4, n = lane & 15, col = nt * 16 + n;
#pragma unroll
  for (int vr = 0; vr < 8; ++vr) {
    int m = vr + half * 8, s = mt * 16 + m;
    if (col < ACTD) {
      float v = acc[vr] + ab2[(long)t * ACTD + col];
      float msk = inputs[((long)b * SS + s) * INW + (GLOB + NA * AFEAT) + col];
      out[((long)b * SS + s) * ACTD + col] = (msk != 0.f) ? v : 0.f;
    }
  }
}

// ================= host =================
static inline long divup(long a, long b) { return (a + b - 1) / b; }

extern "C" void kernel_launch(void* const* d_in, const int* in_sizes, int n_in,
                              void* d_out, int out_size, void* d_ws, size_t ws_size,
                              hipStream_t stream) {
  const float* inputs = (const float*)d_in[0];
  const float* state  = (const float*)d_in[1];
  const int*   ptypes = (const int*)d_in[2];
  const float* emb_W  = (const float*)d_in[3];
  const float* emb_b  = (const float*)d_in[4];
  const float* W_ih   = (const float*)d_in[5];
  const float* W_hh   = (const float*)d_in[6];
  const float* b_ih   = (const float*)d_in[7];
  const float* b_hh   = (const float*)d_in[8];
  const float* W0     = (const float*)d_in[9];
  const float* a0     = (const float*)d_in[10];
  const float* W1     = (const float*)d_in[11];
  const float* a1     = (const float*)d_in[12];
  const float* aW1    = (const float*)d_in[13];
  const float* ab1    = (const float*)d_in[14];
  const float* aW2    = (const float*)d_in[15];
  const float* ab2    = (const float*)d_in[16];
  float* out = (float*)d_out;
  (void)in_sizes; (void)n_in; (void)out_size; (void)ws_size;

  char* ws = (char*)d_ws;
  size_t off = 0;
  auto alloc = [&](size_t bytes) { off = (off + 255) & ~(size_t)255; size_t o = off; off += bytes; return o; };

  __bf16* vec_bf = (__bf16*)(ws + alloc((size_t)ROWS * SS * OBSP * 2));
  __bf16* embWp  = (__bf16*)(ws + alloc(5ULL * OBSP * EMB * 2));
  __bf16* WihP   = (__bf16*)(ws + alloc((size_t)EMB * 3 * HH * 2));
  __bf16* WhhP   = (__bf16*)(ws + alloc((size_t)HH * 3 * HH * 2));
  __bf16* W0P    = (__bf16*)(ws + alloc((size_t)HH * HEADS * G0O * 2));
  __bf16* W1P    = (__bf16*)(ws + alloc((size_t)HH * HEADS * G1O * 2));
  __bf16* aW1p   = (__bf16*)(ws + alloc(5ULL * 512 * ACTH * 2));
  __bf16* aW2p   = (__bf16*)(ws + alloc(5ULL * ACTH * ACTDP * 2));
  __bf16* emb_bf = (__bf16*)(ws + alloc((size_t)ROWS * SS * EMB * 2));
  float*  hf[2]; __bf16* hb[2];
  hf[0] = (float*)(ws + alloc((size_t)ROWS * HH * 4));
  hf[1] = (float*)(ws + alloc((size_t)ROWS * HH * 4));
  hb[0] = (__bf16*)(ws + alloc((size_t)ROWS * HH * 2));
  hb[1] = (__bf16*)(ws + alloc((size_t)ROWS * HH * 2));
  __bf16* gnn    = (__bf16*)(ws + alloc((size_t)BB * SS * NA * HH * 2));  // rnn_out -> g0 (in-place)
  __bf16* feat0  = (__bf16*)(ws + alloc((size_t)BB * SS * HEADS * G1O * 2));
  __bf16* h1     = (__bf16*)(ws + alloc((size_t)BB * SS * ACTH * 2));

  const int T = 256;
  // --- input conversion + B-operand packing into WMMA fragment-tile layout ---
  k_prep_vec<<<divup((long)ROWS * SS * OBSP, T), T, 0, stream>>>(inputs, vec_bf);
  k_pack_embW<<<divup(5L * OBSP * EMB, T), T, 0, stream>>>(emb_W, embWp);
  k_pack_T<<<divup((long)3 * HH * EMB, T), T, 0, stream>>>(W_ih, WihP, 3 * HH, EMB);
  k_pack_T<<<divup((long)3 * HH * HH, T), T, 0, stream>>>(W_hh, WhhP, 3 * HH, HH);
  k_pack_heads<<<divup((long)HEADS * HH * G0O, T), T, 0, stream>>>(W0, W0P, HEADS, HH, G0O);
  k_pack_heads<<<divup((long)HEADS * HH * G1O, T), T, 0, stream>>>(W1, W1P, HEADS, HH, G1O);
  k_pack_aW1<<<divup(5L * 512 * ACTH, T), T, 0, stream>>>(aW1, aW1p);
  k_pack_aW2<<<divup(5L * ACTH * ACTDP, T), T, 0, stream>>>(aW2, aW2p);
  k_init_h<<<divup((long)ROWS * HH, T), T, 0, stream>>>(state, hf[0], hb[0]);

  // --- embedding GEMM ---
  k_embed<<<dim3(ROWS, SS / 16, 8), 128, 0, stream>>>(vec_bf, embWp, emb_b, ptypes, emb_bf);

  // --- GRU recurrence: 512 fused steps ---
  for (int t = 0; t < SS; ++t) {
    int p = t & 1, q = (t + 1) & 1;
    k_gru_step<<<dim3(ROWS / 16, 16), 128, 0, stream>>>(
        emb_bf, WihP, WhhP, b_ih, b_hh, hf[p], hb[p], hf[q], hb[q], gnn, t);
  }
  // final h ends in buffer 0 (512 even steps)
  k_copy_hfinal<<<divup((long)ROWS * HH, T), T, 0, stream>>>(hf[0], out + (long)BB * SS * ACTD);

  // --- GAT layers ---
  k_gat0<<<BB * SS, 256, 0, stream>>>(gnn, W0P, a0);
  k_gat1<<<BB * SS, 256, 0, stream>>>(gnn, W1P, a1, feat0);

  // --- actor head ---
  k_actor1<<<dim3(BB, SS / 16, 4), 128, 0, stream>>>(feat0, aW1p, ab1, ptypes, h1);
  k_actor2<<<dim3(BB, SS / 16), 128, 0, stream>>>(h1, aW2p, ab2, ptypes, inputs, out);
}
